// QJL_27075473834426
// MI455X (gfx1250) — matrix operand order repaired and checked by reading, not compiled, not run on hardware
//
#include <hip/hip_runtime.h>
#include <hip/hip_bf16.h>

typedef __attribute__((ext_vector_type(16))) _Float16 v16h;
typedef __attribute__((ext_vector_type(8)))  _Float16 v8h;
typedef __attribute__((ext_vector_type(8)))  float    v8f;
typedef __attribute__((ext_vector_type(4)))  float    v4f;

#define QJL_D   128   // head dim (= reduction dim of projection)
#define QJL_M   128   // projection dim (= reduction dim of main GEMM)
#define QJL_BQ  4096
#define QJL_BK  8192
#define QJL_SCALE 0.009791516697f   // sqrt(pi/2)/128

// ---------- fragment loaders ----------
// A-matrix (16-bit, 16x32): lane holds row (lane&15); elems 0..15 = K[kb..kb+15],
// where kb = kk*32 + ((lane>=16)?16:0).
__device__ __forceinline__ v16h frag_a_f16(const _Float16* p) {
  union { v16h v; v8h h[2]; } u;
  u.h[0] = *(const v8h*)(p);
  u.h[1] = *(const v8h*)(p + 8);
  return u.v;
}
__device__ __forceinline__ v16h frag_a_f32(const float* p) {
  v4f a0 = *(const v4f*)(p);
  v4f a1 = *(const v4f*)(p + 4);
  v4f b0 = *(const v4f*)(p + 8);
  v4f b1 = *(const v4f*)(p + 12);
  v16h f;
#pragma unroll
  for (int i = 0; i < 4; ++i) {
    f[i]      = (_Float16)a0[i];
    f[4 + i]  = (_Float16)a1[i];
    f[8 + i]  = (_Float16)b0[i];
    f[12 + i] = (_Float16)b1[i];
  }
  return f;
}
// B-matrix (16-bit, 32x16): lane holds column (lane&15); 16 consecutive K
// values starting at kb2 = kk*32 + ((lane>=16)?16:0).
__device__ __forceinline__ v16h frag_b_f16(const _Float16* p) {
  union { v16h v; v8h h[2]; } u;
  u.h[0] = *(const v8h*)(p);
  u.h[1] = *(const v8h*)(p + 8);
  return u.v;
}
__device__ __forceinline__ v16h frag_b_f32(const float* p) {
  v4f a0 = *(const v4f*)(p);
  v4f a1 = *(const v4f*)(p + 4);
  v4f a2 = *(const v4f*)(p + 8);
  v4f a3 = *(const v4f*)(p + 12);
  v16h f;
#pragma unroll
  for (int i = 0; i < 4; ++i) {
    f[i]      = (_Float16)a0[i];
    f[4 + i]  = (_Float16)a1[i];
    f[8 + i]  = (_Float16)a2[i];
    f[12 + i] = (_Float16)a3[i];
  }
  return f;
}

// ---------- kernel 1: C = X @ S^T (R x 128), stored f16; mode 1 -> sign(C) ----------
// One 16x16 tile per wave; 8 waves per block cover all 8 N-tiles of one M-tile.
__global__ __launch_bounds__(256) void qjl_project_kernel(
    const float* __restrict__ X, const float* __restrict__ S,
    _Float16* __restrict__ out16, int mode) {
  const int lane  = threadIdx.x & 31;
  const int wave  = threadIdx.x >> 5;
  const int mBase = blockIdx.x * 16;
  const int nBase = wave * 16;
  const int row   = mBase + (lane & 15);
  const int col   = nBase + (lane & 15);
  const int kselA = (lane >> 4) << 4;   // 0 or 16
  const int kselB = (lane >> 4) << 4;   // 0 or 16

  v8f acc = {};
#pragma unroll
  for (int kk = 0; kk < 4; ++kk) {
    v16h a = frag_a_f32(X + (size_t)row * QJL_D + kk * 32 + kselA);
    v16h b = frag_b_f32(S + (size_t)col * QJL_D + kk * 32 + kselB);
    acc = __builtin_amdgcn_wmma_f32_16x16x32_f16(
        false, a, false, b, (short)0, acc, false, false);
  }
  const int moff = (lane >> 4) << 3;
#pragma unroll
  for (int j = 0; j < 8; ++j) {
    float v = acc[j];
    _Float16 o = mode ? ((v >= 0.0f) ? (_Float16)1.0f : (_Float16)(-1.0f))
                      : (_Float16)v;
    out16[(size_t)(mBase + moff + j) * QJL_M + col] = o;
  }
}

// ---------- kernel 2: colscale[k] = ||residual[k]|| * sqrt(pi/2)/m ----------
__global__ __launch_bounds__(256) void qjl_norm_kernel(
    const float* __restrict__ R, float* __restrict__ colscale) {
  const int lane = threadIdx.x & 31;
  const int wave = threadIdx.x >> 5;
  const int row  = blockIdx.x * 8 + wave;
  const float* p = R + (size_t)row * QJL_D;
  float s = 0.f;
#pragma unroll
  for (int i = 0; i < 4; ++i) {
    float v = p[lane + i * 32];
    s += v * v;
  }
#pragma unroll
  for (int m = 16; m >= 1; m >>= 1) s += __shfl_xor(s, m, 32);
  if (lane == 0) colscale[row] = sqrtf(s) * QJL_SCALE;
}

// ---------- kernel 3: out[q][k] = colscale[k] * (qp @ sg^T) ----------
// Block = 8 waves (2 M x 4 N). Wave tile = 16(M) x 64(N): 1 A-frag reused
// over 4 B-frags per K-chunk; 16 WMMAs per wave. Inputs live in L2 (3 MB).
__global__ __launch_bounds__(256) void qjl_gemm_kernel(
    const _Float16* __restrict__ qp, const _Float16* __restrict__ sg,
    const float* __restrict__ colscale, float* __restrict__ out) {
  const int lane  = threadIdx.x & 31;
  const int wave  = threadIdx.x >> 5;
  const int wm    = wave >> 2;          // 0..1
  const int wn    = wave & 3;           // 0..3
  const int mBase = blockIdx.y * 32 + wm * 16;
  const int nBase = blockIdx.x * 256 + wn * 64;
  const int row   = mBase + (lane & 15);
  const int c0    = nBase + (lane & 15);
  const int kselA = (lane >> 4) << 4;
  const int kselB = (lane >> 4) << 4;

  v8f acc[4] = {v8f{}, v8f{}, v8f{}, v8f{}};
#pragma unroll
  for (int kk = 0; kk < 4; ++kk) {
    const int kbA = kk * 32 + kselA;
    const int kbB = kk * 32 + kselB;
    v16h a = frag_a_f16(qp + (size_t)row * QJL_M + kbA);
#pragma unroll
    for (int t = 0; t < 4; ++t) {
      v16h b = frag_b_f16(sg + (size_t)(c0 + t * 16) * QJL_M + kbB);
      acc[t] = __builtin_amdgcn_wmma_f32_16x16x32_f16(
          false, a, false, b, (short)0, acc[t], false, false);
    }
  }

  const int moff = (lane >> 4) << 3;
#pragma unroll
  for (int t = 0; t < 4; ++t) {
    const int col = c0 + t * 16;
    const float cs = colscale[col];
    float* op = out + (size_t)(mBase + moff) * QJL_BK + col;
#pragma unroll
    for (int j = 0; j < 8; ++j) {
      op[(size_t)j * QJL_BK] = acc[t][j] * cs;
    }
  }
}

extern "C" void kernel_launch(void* const* d_in, const int* in_sizes, int n_in,
                              void* d_out, int out_size, void* d_ws, size_t ws_size,
                              hipStream_t stream) {
  const float* query    = (const float*)d_in[0];   // (BQ, D)
  const float* residual = (const float*)d_in[1];   // (BK, D)
  const float* S        = (const float*)d_in[2];   // (M, D)
  float* out            = (float*)d_out;           // (BQ, BK)

  char* ws = (char*)d_ws;
  _Float16* qp = (_Float16*)(ws);                                        // 1 MB
  _Float16* sg = (_Float16*)(ws + (size_t)QJL_BQ * QJL_M * 2);           // 2 MB
  float* colscale = (float*)(ws + (size_t)(QJL_BQ + QJL_BK) * QJL_M * 2);// 32 KB

  // q_projected (f16) and signs (f16 +/-1)
  qjl_project_kernel<<<QJL_BQ / 16, 256, 0, stream>>>(query, S, qp, 0);
  qjl_project_kernel<<<QJL_BK / 16, 256, 0, stream>>>(residual, S, sg, 1);
  // per-key scale
  qjl_norm_kernel<<<QJL_BK / 8, 256, 0, stream>>>(residual, colscale);
  // main GEMM: 4096 x 8192 x 128
  dim3 grid(QJL_BK / 256, QJL_BQ / 32);
  qjl_gemm_kernel<<<grid, 256, 0, stream>>>(qp, sg, colscale, out);
}
